// KANConv2d_51694226375237
// MI455X (gfx1250) — matrix operand order repaired and checked by reading, not compiled, not run on hardware
//
#include <hip/hip_runtime.h>

typedef _Float16 v8h  __attribute__((ext_vector_type(8)));
typedef _Float16 v16h __attribute__((ext_vector_type(16)));
typedef float    v8f  __attribute__((ext_vector_type(8)));
typedef float    v4f  __attribute__((ext_vector_type(4)));
typedef unsigned int u32x4 __attribute__((ext_vector_type(4)));
typedef int          i32x4 __attribute__((ext_vector_type(4)));
typedef int          i32x8 __attribute__((ext_vector_type(8)));

#define KAUG        9216   // 576 features * 16 (silu + 8 bases + 7 pad)
#define LDS_STRIDE  40     // halves per row: 32 data + 8 pad -> 80B stride, bank-friendly
#define TOK_PER_BLK 64
#define NCHUNK      288    // 576 features / 2 per 32-K chunk
#define A_HALVES    (TOK_PER_BLK * LDS_STRIDE)   // 5120 B per buffer
#define B_HALVES    (128 * LDS_STRIDE)           // 10240 B per buffer

// ---------------------------------------------------------------------------
// Kernel 1: pack [base_weight | scaler*spline_weight | zeros] into f16,
// layout Wp[o][d*16 + j], contiguous K per output channel.
// ---------------------------------------------------------------------------
__global__ __launch_bounds__(256) void pack_weights(const float* __restrict__ bw,
                                                    const float* __restrict__ sw,
                                                    const float* __restrict__ ss,
                                                    _Float16* __restrict__ wp) {
    int idx = blockIdx.x * 256 + threadIdx.x;      // o*576 + d
    if (idx >= 128 * 576) return;
    float scal = ss[idx];
    v8h lo, hi;
    lo[0] = (_Float16)bw[idx];
#pragma unroll
    for (int k = 0; k < 7; ++k) lo[k + 1] = (_Float16)(sw[idx * 8 + k] * scal);
    hi[0] = (_Float16)(sw[idx * 8 + 7] * scal);
#pragma unroll
    for (int k = 1; k < 8; ++k) hi[k] = (_Float16)0.f;
    v8h* dst = (v8h*)(wp + (size_t)idx * 16);
    dst[0] = lo;
    dst[1] = hi;
}

// ---------------------------------------------------------------------------
// Kernel 2: fused im2col + silu + cubic-B-spline + WMMA GEMM with TDM-staged
// weights. 128 threads (4 waves), 64 tokens/block, 128 output channels.
// ---------------------------------------------------------------------------
__global__ __launch_bounds__(128) void kan_wmma(const float* __restrict__ x,
                                                const _Float16* __restrict__ wp,
                                                float* __restrict__ out) {
    __shared__ _Float16 aTile[2][A_HALVES];   // activations, double buffered
    __shared__ _Float16 bTile[2][B_HALVES];   // weights (TDM dest), double buffered

    const int tid     = threadIdx.x;
    const int lane    = tid & 31;
    const int wv      = tid >> 5;          // wave 0..3, owns token rows wv*16..+15
    const int tl      = tid & 63;          // local token for elementwise phase
    const int halfSel = tid >> 6;          // which of the 2 features in this chunk
    const int tok0    = blockIdx.x * TOK_PER_BLK;

    // token -> (b, ho, wo)
    const int n     = tok0 + tl;
    const int bimg  = n >> 12;
    const int rem   = n & 4095;
    const int ho    = rem >> 6, wo = rem & 63;
    const int xbase = bimg << 18;          // (b*64)<<12

    // A-fragment offsets (contiguous K: lane half h holds k 16h..16h+15)
    const int arow = wv * 16 + (lane & 15);
    const int alo  = (lane >> 4) * 16;     // halves: 0 or 16
    // B-fragment: lane = column in 16-wide tile, half-wave selects K half
    const int brow = lane & 15;
    const int bk   = (lane >> 4) * 16;

    // incrementally-maintained feature decomposition: d = 2*chunk + halfSel
    int cch = 0, r9 = halfSel;

    // ---- elementwise producer: silu + cardinal cubic B-spline -> LDS f16 ----
    auto produce = [&](int slot) {
        int kh = (r9 >= 6) ? 2 : ((r9 >= 3) ? 1 : 0);
        int kw = r9 - kh * 3;
        int ih = ho + kh - 1, iw = wo + kw - 1;
        float v = 0.f;
        if ((unsigned)ih < 64u && (unsigned)iw < 64u)
            v = x[xbase + (cch << 12) + (ih << 6) + iw];

        float e    = __expf(-v);
        float silu = v * __builtin_amdgcn_rcpf(1.f + e);

        float uf = (v + 2.2f) * 2.5f;      // grid [-2.2,2.2], h=0.4
        float jf = floorf(uf);
        int   j  = (int)jf;
        float u  = uf - jf;
        float um = 1.f - u;
        float u2 = u * u, u3 = u2 * u;
        const float s = 1.f / 6.f;
        float w0 = um * um * um * s;
        float w1 = (3.f * u3 - 6.f * u2 + 4.f) * s;
        float w2 = (-3.f * u3 + 3.f * u2 + 3.f * u + 1.f) * s;
        float w3 = u3 * s;

        _Float16 hv[16];
#pragma unroll
        for (int k = 0; k < 16; ++k) hv[k] = (_Float16)0.f;
        hv[0] = (_Float16)silu;
#pragma unroll
        for (int k = 0; k < 8; ++k) {
            int   i   = k - j + 3;         // boundary truncation automatic
            float wvv = (i == 0) ? w0 : (i == 1) ? w1 : (i == 2) ? w2
                      : (i == 3) ? w3 : 0.f;
            hv[k + 1] = (_Float16)wvv;
        }
        v8h lo, hi;
#pragma unroll
        for (int k = 0; k < 8; ++k) { lo[k] = hv[k]; hi[k] = hv[k + 8]; }
        v8h* dst = (v8h*)&aTile[slot][tl * LDS_STRIDE + halfSel * 16];
        dst[0] = lo;
        dst[1] = hi;
        // advance feature state by 2
        r9 += 2;
        if (r9 >= 9) { r9 -= 9; ++cch; }
    };

    // ---- TDM: DMA one 32x128 f16 weight slice into LDS (80B padded rows) ----
    auto issueB = [&](int chunk) {
        unsigned long long ga =
            (unsigned long long)(uintptr_t)wp + (unsigned long long)chunk * 64ull;
        unsigned ldsb = (unsigned)(uintptr_t)&bTile[chunk & 1][0];
        u32x4 g0 = { 1u,                                   // count=1 (valid D#)
                     ldsb,                                 // lds_addr
                     (unsigned)ga,                         // global_addr[31:0]
                     (unsigned)((ga >> 32) & 0x1FFFFFFull) // global_addr[56:32]
                         | 0x80000000u };                  // type=2 (image)
        i32x8 g1 = { (int)((1u << 16)    // data_size = 2 bytes
                         | (1u << 20)    // pad_enable
                         | (3u << 22)    // pad_interval: every 16 DWORDs
                         | (3u << 25)),  // pad_amount: 4 DWORDs (16B) -> 80B stride
                     (int)(9216u << 16),   // tensor_dim0 = 9216 (K length)
                     (int)(128u << 16),    // tensor_dim1 = 128 rows
                     (int)(32u << 16),     // tile_dim0 = 32 halves
                     128,                  // tile_dim1 = 128 rows
                     9216,                 // tensor_dim0_stride = 9216
                     0, 0 };
        i32x4 gz = { 0, 0, 0, 0 };
#if __clang_major__ >= 23
        i32x8 z8 = {};
        __builtin_amdgcn_tensor_load_to_lds(g0, g1, gz, gz, z8, 0);
#else
        __builtin_amdgcn_tensor_load_to_lds(g0, g1, gz, gz, 0);
#endif
    };

    v8f acc[8];
#pragma unroll
    for (int c = 0; c < 8; ++c) acc[c] = (v8f){};

    // ---- one pipeline stage: consume chunk kc, stage kc+1 ----
    auto doChunk = [&](int kc, bool stageNext) {
        __builtin_amdgcn_s_wait_tensorcnt(0);   // B(kc) landed (wave0; NOP others)
        __syncthreads();                        // A(kc) visible; old buffers free

        if (stageNext) {
            if (__builtin_amdgcn_readfirstlane(tid) == 0) issueB(kc + 1);
        }

        // ---- A fragment ----
        const _Float16* ap = &aTile[kc & 1][arow * LDS_STRIDE + alo];
        v8h a_lo = *(const v8h*)(ap);
        v8h a_hi = *(const v8h*)(ap + 8);
        v16h afrag = __builtin_shufflevector(a_lo, a_hi,
            0,1,2,3,4,5,6,7,8,9,10,11,12,13,14,15);

        // ---- B fragments: explicit 2-deep ping-pong so load(c+1) is in
        //      flight (different regs, no WAR) while wmma(c) executes ----
        const _Float16* bslot = &bTile[kc & 1][0];
        auto loadB = [&](int c) -> v16h {
            const v8h* bp = (const v8h*)(bslot + (c * 16 + brow) * LDS_STRIDE + bk);
            return __builtin_shufflevector(bp[0], bp[1],
                0,1,2,3,4,5,6,7,8,9,10,11,12,13,14,15);
        };

        v16h bcur = loadB(0);
#pragma unroll
        for (int c = 0; c < 8; ++c) {
            v16h bnext;
            if (c < 7) bnext = loadB(c + 1);        // issue before consuming bcur
            acc[c] = __builtin_amdgcn_wmma_f32_16x16x32_f16(
                false, afrag, false, bcur, (short)0, acc[c], false, false);
            if (c < 7) bcur = bnext;
        }

        // ---- overlap next chunk's elementwise math with the matrix pipe ----
        if (stageNext) produce((kc + 1) & 1);
    };

    // ---- prologue: stage chunk 0 ----
    produce(0);
    if (__builtin_amdgcn_readfirstlane(tid) == 0) issueB(0);

    for (int kc = 0; kc < NCHUNK - 1; ++kc) doChunk(kc, true);
    doChunk(NCHUNK - 1, false);             // peeled epilogue: no staging branch

    // ---- store: C-tile VGPR index == consecutive tokens == contiguous memory ----
    const int tokb = tok0 + wv * 16 + (lane >> 4) * 8;
    const int oc0  = lane & 15;
    const size_t obase = (size_t)((tok0 >> 12) * 128) * 4096 + (size_t)(tokb & 4095);
#pragma unroll
    for (int c = 0; c < 8; ++c) {
        int o = c * 16 + oc0;
        float* p = out + obase + (size_t)o * 4096;
        v4f lo4 = {acc[c][0], acc[c][1], acc[c][2], acc[c][3]};
        v4f hi4 = {acc[c][4], acc[c][5], acc[c][6], acc[c][7]};
        *(v4f*)p       = lo4;
        *(v4f*)(p + 4) = hi4;
    }
}

// ---------------------------------------------------------------------------
extern "C" void kernel_launch(void* const* d_in, const int* in_sizes, int n_in,
                              void* d_out, int out_size, void* d_ws, size_t ws_size,
                              hipStream_t stream) {
    const float* x  = (const float*)d_in[0];   // (4,64,64,64)
    const float* bw = (const float*)d_in[1];   // (128,576)
    const float* sw = (const float*)d_in[2];   // (128,576,8)
    const float* ss = (const float*)d_in[3];   // (128,576)
    float* out = (float*)d_out;                // (4,128,64,64)

    _Float16* wp = (_Float16*)d_ws;            // needs 128*9216*2 = 2,359,296 B

    pack_weights<<<(128 * 576 + 255) / 256, 256, 0, stream>>>(bw, sw, ss, wp);
    kan_wmma<<<16384 / TOK_PER_BLK, 128, 0, stream>>>(x, wp, out);
}
